// FractalAttention_19868518711386
// MI455X (gfx1250) — compile-verified, matrix operand order checked by reading, not run on hardware
//
#include <hip/hip_runtime.h>
#include <stdint.h>

// ---------------------------------------------------------------------------
// CDNA5 (gfx1250) bf16-WMMA implementation of multi-head attention.
// - All matrix math via v_wmma_f32_16x16x32_bf16 (f32 accumulate).
// - GEMMs and attention stage operand tiles global->LDS with the CDNA5 async
//   copy engine (ASYNCcnt), double buffered; fragments served by ds_load_b128.
// ---------------------------------------------------------------------------

typedef __bf16 bf16_t;
typedef __attribute__((ext_vector_type(8)))  __bf16 v8bf;
typedef __attribute__((ext_vector_type(16))) __bf16 v16bf;
typedef __attribute__((ext_vector_type(8)))  float  v8f;

#define T_SEQ     2048
#define D_MODEL   2048
#define NUM_HEADS 16
#define HEAD_DIM  128
#define B_SZ      4
#define M_ROWS    (B_SZ * T_SEQ)   // 8192

#define KCHUNK    64               // GEMM k-slab staged to LDS
#define APAD      72               // padded GEMM LDS row (elems): 144B
#define KPAD      136              // attn K-tile row pad (128+8 elems): 272B
#define VPAD      40               // attn V-tile row pad (32+8 elems): 80B

// Load a 16-element bf16 A/B fragment slice for one lane.
// Per ISA 7.12.2 (16-bit 16x32): lane-half selects K-subgroup; elements
// 0..7 come from [p, p+8), elements 8..15 from [p+16, p+24). Caller passes
// p = row_base + k_chunk + half*8 (16B aligned). Works on global or LDS ptrs.
__device__ __forceinline__ v16bf load_frag(const bf16_t* p) {
  v8bf lo = *(const v8bf*)(p);
  v8bf hi = *(const v8bf*)(p + 16);
  v16bf r;
#pragma unroll
  for (int i = 0; i < 8; ++i) { r[i] = lo[i]; r[i + 8] = hi[i]; }
  return r;
}

__device__ __forceinline__ v8f wmma_bf16(v16bf a, v16bf b, v8f c) {
  // D = A(16x32) * B(32x16) + C, f32 accumulate.
  return __builtin_amdgcn_wmma_f32_16x16x32_bf16(
      /*neg_a=*/false, a, /*neg_b=*/false, b,
      /*c_mod=*/(short)0, c, /*reuse_a=*/false, /*reuse_b=*/false);
}

// Async 16-byte global->LDS copy (per-lane), tracked by ASYNCcnt.
// lds_addr = byte address within LDS (generic pointer truncated to 32 bits,
// ISA sec 10.2: LDS aperture addrs map to LDS via addr[31:0]).
__device__ __forceinline__ void async_load_b128(uint32_t lds_addr,
                                                const bf16_t* gptr) {
  asm volatile("global_load_async_to_lds_b128 %0, %1, off"
               :: "v"(lds_addr), "v"(gptr)
               : "memory");
}

__device__ __forceinline__ void wait_async_all() {
  asm volatile("s_wait_asynccnt 0x0" ::: "memory");
}

// ---------------------------------------------------------------------------
// Elementwise f32 -> bf16 cast (grid-stride).
// ---------------------------------------------------------------------------
__global__ void cast_f32_to_bf16(const float* __restrict__ in,
                                 bf16_t* __restrict__ out, size_t n) {
  size_t i = (size_t)blockIdx.x * blockDim.x + threadIdx.x;
  size_t stride = (size_t)gridDim.x * blockDim.x;
  for (; i < n; i += stride) out[i] = (bf16_t)in[i];
}

// ---------------------------------------------------------------------------
// GEMM: out = A(M x K, bf16 row-major) * W^T (W is N x K bf16 row-major) + bias
// Block tile 128(M) x 128(N), 8 waves = 4(M) x 2(N), each 32x64 (2x4 WMMA).
// K staged in 64-deep slabs to LDS via async copies, double buffered.
// mode 0: bf16 out, (B,H,T,hd) head layout (for Q,K)
// mode 1: bf16 out, (B,H,hd,T) transposed   (for V)
// mode 2: f32 out, row-major M x N          (final projection)
// ---------------------------------------------------------------------------
__global__ __launch_bounds__(256) void gemm_bf16_wmma(
    const bf16_t* __restrict__ A, const bf16_t* __restrict__ W,
    const float* __restrict__ bias, void* __restrict__ outp,
    int Kdim, int N, int mode) {
  const int tid  = threadIdx.x;
  const int wave = tid >> 5;
  const int lane = tid & 31;
  const int half = lane >> 4;
  const int ln   = lane & 15;
  const int mw = wave & 3;            // 4 waves along M (32 rows each)
  const int nw = wave >> 2;           // 2 waves along N (64 cols each)
  const int mbase = blockIdx.y * 128 + mw * 32;
  const int nbase = blockIdx.x * 128 + nw * 64;
  const int mblk  = blockIdx.y * 128;
  const int nblk  = blockIdx.x * 128;

  // Double-buffered LDS tiles: A 128x64, W 128x64 (rows padded to 72 elems).
  __shared__ bf16_t smemA[2][128 * APAD];
  __shared__ bf16_t smemW[2][128 * APAD];

  // Stage one 64-deep k-slab: each tile is 128 rows x 8 b128-chunks
  // (1024 copies); 256 threads -> 4 chunks per thread per tile.
  auto stage = [&](int k0, int buf) {
#pragma unroll
    for (int i = 0; i < 4; ++i) {
      const int cid = tid + i * 256;        // 0..1023
      const int row = cid >> 3, c = cid & 7;
      async_load_b128((uint32_t)(uintptr_t)(&smemA[buf][row * APAD + c * 8]),
                      A + (size_t)(mblk + row) * Kdim + k0 + c * 8);
      async_load_b128((uint32_t)(uintptr_t)(&smemW[buf][row * APAD + c * 8]),
                      W + (size_t)(nblk + row) * Kdim + k0 + c * 8);
    }
  };

  v8f acc[2][4] = {};

  stage(0, 0);
  int cur = 0;
  for (int k0 = 0; k0 < Kdim; k0 += KCHUNK) {
    wait_async_all();        // my staged copies for buf[cur] have landed
    __syncthreads();         // everyone's copies visible to all waves
    if (k0 + KCHUNK < Kdim) stage(k0 + KCHUNK, cur ^ 1);

    const bf16_t* aBuf = smemA[cur];
    const bf16_t* wBuf = smemW[cur];
#pragma unroll
    for (int kk = 0; kk < KCHUNK; kk += 32) {
      const int off = kk + half * 8;
      v16bf a0 = load_frag(aBuf + (mw * 32 + ln)      * APAD + off);
      v16bf a1 = load_frag(aBuf + (mw * 32 + 16 + ln) * APAD + off);
#pragma unroll
      for (int j = 0; j < 4; ++j) {
        v16bf bj = load_frag(wBuf + (nw * 64 + j * 16 + ln) * APAD + off);
        acc[0][j] = wmma_bf16(a0, bj, acc[0][j]);
        acc[1][j] = wmma_bf16(a1, bj, acc[1][j]);
      }
    }
    __syncthreads();         // all reads of buf[cur] done before it is reused
    cur ^= 1;
  }

  // C/D layout: vgpr g, lane -> row = g + 8*half, col = ln (ISA 7.12.2)
#pragma unroll
  for (int i = 0; i < 2; ++i)
#pragma unroll
    for (int j = 0; j < 4; ++j)
#pragma unroll
      for (int g = 0; g < 8; ++g) {
        const int m = mbase + i * 16 + g + 8 * half;
        const int n = nbase + j * 16 + ln;
        const float v = acc[i][j][g] + bias[n];
        if (mode == 2) {
          ((float*)outp)[(size_t)m * N + n] = v;
        } else {
          const int bb = m >> 11, tt = m & (T_SEQ - 1);
          const int hh = n >> 7,  dd = n & (HEAD_DIM - 1);
          size_t idx;
          if (mode == 0)
            idx = ((size_t)(bb * NUM_HEADS + hh) * T_SEQ + tt) * HEAD_DIM + dd;
          else
            idx = ((size_t)(bb * NUM_HEADS + hh) * HEAD_DIM + dd) * T_SEQ + tt;
          ((bf16_t*)outp)[idx] = (bf16_t)v;
        }
      }
}

// ---------------------------------------------------------------------------
// Flash attention: block = one (b,h) x 128 queries; 8 waves x 16 queries.
// Online softmax over key chunks of 32. K/V tiles for the chunk are staged
// into double-buffered LDS with async copies and shared by all 8 waves.
// Q,K in (B,H,T,hd); V transposed (B,H,hd,T); output Ao in (B,T,H*hd) bf16.
// ---------------------------------------------------------------------------
__global__ __launch_bounds__(256) void attn_fa_kernel(
    const bf16_t* __restrict__ Qh, const bf16_t* __restrict__ Kh,
    const bf16_t* __restrict__ Vt, const int* __restrict__ mask,
    bf16_t* __restrict__ Ao) {
  const int bh = blockIdx.x;
  const int b  = bh / NUM_HEADS;
  const int h  = bh % NUM_HEADS;
  const int tid  = threadIdx.x;
  const int wave = tid >> 5;
  const int lane = tid & 31;
  const int half = lane >> 4;
  const int ln   = lane & 15;
  const int q0   = blockIdx.y * 128 + wave * 16;   // 16 queries per wave

  const bf16_t* Qrow  = Qh + ((size_t)bh * T_SEQ + q0 + ln) * HEAD_DIM;
  const bf16_t* Kbase = Kh + (size_t)bh * T_SEQ * HEAD_DIM;
  const bf16_t* Vbase = Vt + (size_t)bh * HEAD_DIM * T_SEQ;
  const int* mask_b   = mask + b * T_SEQ;

  // Double-buffered K tile (32 keys x 128 d) and V tile (128 d x 32 keys).
  __shared__ bf16_t kls[2][32 * KPAD];    // 17,408 B
  __shared__ bf16_t vls[2][128 * VPAD];   // 20,480 B
  // Wave-private P tile (16 queries x 32 keys), padded.
  __shared__ bf16_t plds[8][16][40];      // 10,240 B

  // Stage K tile (32 rows x 16 b128-chunks = 512) and V tile (128 rows x 4
  // chunks = 512): 2 + 2 async copies per thread per chunk.
  auto stage_kv = [&](int k0, int buf) {
#pragma unroll
    for (int i = 0; i < 2; ++i) {
      const int cid = tid + i * 256;            // 0..511
      const int krow = cid >> 4, kc = cid & 15;
      async_load_b128((uint32_t)(uintptr_t)(&kls[buf][krow * KPAD + kc * 8]),
                      Kbase + (size_t)(k0 + krow) * HEAD_DIM + kc * 8);
      const int vrow = cid >> 2, vc = cid & 3;
      async_load_b128((uint32_t)(uintptr_t)(&vls[buf][vrow * VPAD + vc * 8]),
                      Vbase + (size_t)vrow * T_SEQ + k0 + vc * 8);
    }
  };

  // Preload Q A-fragments for all 4 d-chunks (hd = 128 = 4 * 32).
  v16bf qf[4];
#pragma unroll
  for (int c = 0; c < 4; ++c)
    qf[c] = load_frag(Qrow + c * 32 + half * 8);

  float rmax[8], rsum[8];
#pragma unroll
  for (int g = 0; g < 8; ++g) { rmax[g] = -3.0e38f; rsum[g] = 0.f; }
  v8f acc[8] = {};    // 8 d-tiles of 16x16 => hd = 128

  const float scale = 0.08838834764831845f;   // 1/sqrt(128)

  stage_kv(0, 0);
  int cur = 0;
  for (int k0 = 0; k0 < T_SEQ; k0 += 32) {
    wait_async_all();
    __syncthreads();
    if (k0 + 32 < T_SEQ) stage_kv(k0 + 32, cur ^ 1);
    const bf16_t* kb = kls[cur];
    const bf16_t* vb = vls[cur];

    // ---- scores: two 16-key tiles, S = Q * K^T (B column = K-tile row) ----
    v8f s[2];
#pragma unroll
    for (int t = 0; t < 2; ++t) {
      v8f c = {};
#pragma unroll
      for (int c4 = 0; c4 < 4; ++c4) {
        v16bf kf = load_frag(kb + (t * 16 + ln) * KPAD + c4 * 32 + half * 8);
        c = wmma_bf16(qf[c4], kf, c);
      }
      const int mv = mask_b[k0 + t * 16 + ln];
#pragma unroll
      for (int g = 0; g < 8; ++g)
        s[t][g] = (mv == 0) ? -3.0e38f : c[g] * scale;
    }

    // ---- online softmax; row r = g + 8*half lives across 16 lanes of a half ----
#pragma unroll
    for (int g = 0; g < 8; ++g) {
      float v = fmaxf(s[0][g], s[1][g]);
      v = fmaxf(v, __shfl_xor(v, 1, 32));
      v = fmaxf(v, __shfl_xor(v, 2, 32));
      v = fmaxf(v, __shfl_xor(v, 4, 32));
      v = fmaxf(v, __shfl_xor(v, 8, 32));
      const float mnew = fmaxf(rmax[g], v);
      const float corr = __expf(rmax[g] - mnew);
      rmax[g] = mnew;
      const float p0 = __expf(s[0][g] - mnew);
      const float p1 = __expf(s[1][g] - mnew);
      float rs = p0 + p1;
      rs += __shfl_xor(rs, 1, 32);
      rs += __shfl_xor(rs, 2, 32);
      rs += __shfl_xor(rs, 4, 32);
      rs += __shfl_xor(rs, 8, 32);
      rsum[g] = rsum[g] * corr + rs;
#pragma unroll
      for (int dt = 0; dt < 8; ++dt) acc[dt][g] *= corr;
      const int r = g + 8 * half;
      plds[wave][r][ln]      = (bf16_t)p0;
      plds[wave][r][16 + ln] = (bf16_t)p1;
    }
    // Wave-private LDS round-trip: per-wave DS ops are hardware in-order;
    // the clobber stops compiler reordering of the cross-lane traffic.
    asm volatile("" ::: "memory");

    // ---- P as A-fragment (16 queries x 32 keys) ----
    v16bf pf = load_frag(&plds[wave][ln][half * 8]);

    // ---- acc += P * V (B column d = V-tile row d, contiguous over keys) ----
#pragma unroll
    for (int dt = 0; dt < 8; ++dt) {
      v16bf vf = load_frag(vb + (dt * 16 + ln) * VPAD + half * 8);
      acc[dt] = wmma_bf16(pf, vf, acc[dt]);
    }
    __syncthreads();   // all reads of buf[cur] done before it is overwritten
    cur ^= 1;
  }

  // ---- normalize and store to (B, T, H*hd) bf16 ----
#pragma unroll
  for (int dt = 0; dt < 8; ++dt)
#pragma unroll
    for (int g = 0; g < 8; ++g) {
      const int t = q0 + g + 8 * half;
      const int d = dt * 16 + ln;
      const float o = acc[dt][g] / rsum[g];
      Ao[((size_t)(b * T_SEQ + t)) * D_MODEL + h * HEAD_DIM + d] = (bf16_t)o;
    }
}

// ---------------------------------------------------------------------------
// Host-side orchestration.
// ---------------------------------------------------------------------------
extern "C" void kernel_launch(void* const* d_in, const int* in_sizes, int n_in,
                              void* d_out, int out_size, void* d_ws, size_t ws_size,
                              hipStream_t stream) {
  const float* x   = (const float*)d_in[0];
  const int*   msk = (const int*)d_in[1];
  const float* Wq  = (const float*)d_in[2];
  const float* bq  = (const float*)d_in[3];
  const float* Wk  = (const float*)d_in[4];
  const float* bk  = (const float*)d_in[5];
  const float* Wv  = (const float*)d_in[6];
  const float* bv  = (const float*)d_in[7];
  const float* Wo  = (const float*)d_in[8];
  const float* bo  = (const float*)d_in[9];
  float* out = (float*)d_out;

  const size_t nX = (size_t)M_ROWS * D_MODEL;    // 16,777,216
  const size_t nW = (size_t)D_MODEL * D_MODEL;   //  4,194,304

  bf16_t* p   = (bf16_t*)d_ws;                   // ~192 MiB total bf16 scratch
  bf16_t* xb  = p; p += nX;
  bf16_t* wqb = p; p += nW;
  bf16_t* wkb = p; p += nW;
  bf16_t* wvb = p; p += nW;
  bf16_t* wob = p; p += nW;
  bf16_t* Qh  = p; p += nX;   // (B,H,T,hd)
  bf16_t* Kh  = p; p += nX;   // (B,H,T,hd)
  bf16_t* Vth = p; p += nX;   // (B,H,hd,T)
  bf16_t* Ao  = p; p += nX;   // (B,T,D)

  cast_f32_to_bf16<<<4096, 256, 0, stream>>>(x,  xb,  nX);
  cast_f32_to_bf16<<<1024, 256, 0, stream>>>(Wq, wqb, nW);
  cast_f32_to_bf16<<<1024, 256, 0, stream>>>(Wk, wkb, nW);
  cast_f32_to_bf16<<<1024, 256, 0, stream>>>(Wv, wvb, nW);
  cast_f32_to_bf16<<<1024, 256, 0, stream>>>(Wo, wob, nW);

  dim3 gg(D_MODEL / 128, M_ROWS / 128);          // (16, 64)
  gemm_bf16_wmma<<<gg, 256, 0, stream>>>(xb, wqb, bq, Qh,  D_MODEL, D_MODEL, 0);
  gemm_bf16_wmma<<<gg, 256, 0, stream>>>(xb, wkb, bk, Kh,  D_MODEL, D_MODEL, 0);
  gemm_bf16_wmma<<<gg, 256, 0, stream>>>(xb, wvb, bv, Vth, D_MODEL, D_MODEL, 1);

  attn_fa_kernel<<<dim3(B_SZ * NUM_HEADS, T_SEQ / 128), 256, 0, stream>>>(
      Qh, Kh, Vth, msk, Ao);

  gemm_bf16_wmma<<<gg, 256, 0, stream>>>(Ao, wob, bo, out, D_MODEL, D_MODEL, 2);
}
